// AlgoGAT_75780402971086
// MI455X (gfx1250) — matrix operand order repaired and checked by reading, not compiled, not run on hardware
//
#include <hip/hip_runtime.h>
#include <hip/hip_bf16.h>
#include <math.h>

// ---------------------------------------------------------------------------
// GATv2 actor-critic forward for MI455X (gfx1250, wave32, WMMA).
// All GEMMs: bf16 operands pre-converted/pre-transposed once, then a clean
// v_wmma_f32_16x16x32_bf16 inner loop (2 A loads + 8 B loads + 8 WMMAs per k-step,
// 16x128 row strip per wave). Edge softmax/scatter kept fp32 (HBM-bound).
// ---------------------------------------------------------------------------

typedef __attribute__((ext_vector_type(8)))  __bf16 v8bf;
typedef __attribute__((ext_vector_type(16))) __bf16 v16bf;
typedef __attribute__((ext_vector_type(8)))  float  v8f;

#define HD      128     // hidden
#define NNODES  32768
#define NEDGES  524288
#define NGRAPH  16
#define NF      64
#define EF      32
#define NLAYERS 3
#define SLOPE   0.2f

__device__ __forceinline__ __bf16 f2bf(float f) {
  unsigned u = __builtin_bit_cast(unsigned, f);
  unsigned r = u + 0x7FFFu + ((u >> 16) & 1u);   // RNE to bf16
  unsigned short h = (unsigned short)(r >> 16);
  return __builtin_bit_cast(__bf16, h);
}
// order-preserving float<->uint mapping for atomic segment-max
__device__ __forceinline__ unsigned f2ord(float f) {
  unsigned u = __builtin_bit_cast(unsigned, f);
  return (u & 0x80000000u) ? ~u : (u | 0x80000000u);
}
__device__ __forceinline__ float ord2f(unsigned u) {
  unsigned b = (u & 0x80000000u) ? (u & 0x7FFFFFFFu) : ~u;
  return __builtin_bit_cast(float, b);
}

// ---------------------------------------------------------------------------
// GEMM: out[M,128] = act(A[M,K] @ W[K,128] + bias)
//   A  : bf16 row-major  (pre-converted)
//   Wt : bf16 [128,K]    (pre-transposed weights; B-fragment = 16 contiguous
//        bf16 at Wt[n*K + kt + khalf*16], same k order as A)
//   A-fragment: elements 0..7  = A[m][kt+khalf*16 .. +7]
//               elements 8..15 = A[m][kt+khalf*16+8 .. +7]   (two 16B loads)
// One wave computes a 16x128 row strip (8 output tiles, A reused 8x).
// K is a template param so the k-loop fully unrolls around the WMMAs.
// ---------------------------------------------------------------------------
template <int K>
__global__ void __launch_bounds__(256)
wmma_gemm_bf16(const __bf16* __restrict__ A, const __bf16* __restrict__ Wt,
               const float* __restrict__ bias, float* __restrict__ out,
               int act) {
  const int l     = threadIdx.x & 31;
  const int wave  = threadIdx.x >> 5;
  const int waves = blockDim.x >> 5;
  const int tm    = blockIdx.x * waves + wave;   // 16-row strip index
  const int khalf = l >> 4;
  const int n     = l & 15;

  const __bf16* Arow = A + (size_t)(tm * 16 + (l & 15)) * K + khalf * 16;

  v8f acc[8] = {};
#pragma unroll
  for (int kt = 0; kt < K; kt += 32) {
    const v8bf alo = *(const v8bf*)(Arow + kt);
    const v8bf ahi = *(const v8bf*)(Arow + kt + 8);
    const v16bf a = __builtin_shufflevector(
        alo, ahi, 0, 1, 2, 3, 4, 5, 6, 7, 8, 9, 10, 11, 12, 13, 14, 15);
#pragma unroll
    for (int t = 0; t < 8; ++t) {
      const v16bf b =
          *(const v16bf*)(Wt + (size_t)(t * 16 + n) * K + kt + khalf * 16);
      acc[t] = __builtin_amdgcn_wmma_f32_16x16x32_bf16(
          false, a, false, b, (short)0, acc[t], false, false);
    }
  }

#pragma unroll
  for (int t = 0; t < 8; ++t) {
    const int ncol = t * 16 + n;
    const float bv = bias ? bias[ncol] : 0.0f;
    float* ocol = out + (size_t)(tm * 16 + khalf * 8) * HD + ncol;
#pragma unroll
    for (int v = 0; v < 8; ++v) {
      float val = acc[t][v] + bv;
      if (act) val = fmaxf(val, 0.0f);
      ocol[(size_t)v * HD] = val;
    }
  }
}

// ---------------------------------------------------------------------------
__global__ void f2bf_kernel(const float* __restrict__ in, __bf16* __restrict__ out,
                            long long n) {
  long long i = (long long)blockIdx.x * blockDim.x + threadIdx.x;
  const long long stride = (long long)gridDim.x * blockDim.x;
  for (; i < n; i += stride) out[i] = f2bf(in[i]);
}

// W [K,128] f32 -> Wt [128,K] bf16
__global__ void wtrans_kernel(const float* __restrict__ W, __bf16* __restrict__ Wt,
                              int K) {
  const int idx = blockIdx.x * blockDim.x + threadIdx.x;
  if (idx >= K * HD) return;
  const int k = idx >> 7, nn = idx & (HD - 1);
  Wt[(size_t)nn * K + k] = f2bf(W[(size_t)k * HD + nn]);
}

__global__ void fill_kernel(float* p, float v, long long n) {
  long long i = (long long)blockIdx.x * blockDim.x + threadIdx.x;
  const long long stride = (long long)gridDim.x * blockDim.x;
  for (; i < n; i += stride) p[i] = v;
}

// ---------------------------------------------------------------------------
// Edge pass 1: m = leaky_relu(xl[src]+xr[dst]+xe), logit[h] = dot(m_h, att_h);
// store logit, atomic segment-max over dst.  One wave per edge; lane owns 4
// channels (lanes 0-15 -> head 0, lanes 16-31 -> head 1).
// ---------------------------------------------------------------------------
__global__ void edge_logit_kernel(const float* __restrict__ xl,
                                  const float* __restrict__ xr,
                                  const float* __restrict__ xe,
                                  const int* __restrict__ src,
                                  const int* __restrict__ dst,
                                  const float* __restrict__ att,   // [2*64] flat
                                  float* __restrict__ logit,       // [E,2]
                                  unsigned* __restrict__ mx,       // [N,2] ord
                                  int E) {
  const int e = blockIdx.x * (blockDim.x >> 5) + (threadIdx.x >> 5);
  const int l = threadIdx.x & 31;
  if (e >= E) return;
  const int s = src[e], d = dst[e];
  const int cbase = l * 4;
  float part = 0.0f;
#pragma unroll
  for (int j = 0; j < 4; ++j) {
    const int c = cbase + j;
    float m = xl[(size_t)s * HD + c] + xr[(size_t)d * HD + c] +
              xe[(size_t)e * HD + c];
    m = (m > 0.0f) ? m : SLOPE * m;
    part += m * att[c];
  }
  for (int off = 1; off < 16; off <<= 1) part += __shfl_xor(part, off, 32);
  if (l == 0 || l == 16) {
    const int h = l >> 4;
    logit[(size_t)e * 2 + h] = part;
    atomicMax(&mx[(size_t)d * 2 + h], f2ord(part));
  }
}

// Edge pass 2: ex = exp(logit - mx[dst]); segment-sum into den.
__global__ void edge_exp_kernel(const float* __restrict__ logit,
                                const int* __restrict__ dst,
                                const unsigned* __restrict__ mx,
                                float* __restrict__ exb,
                                float* __restrict__ den, int E2) {
  const int idx = blockIdx.x * blockDim.x + threadIdx.x;
  if (idx >= E2) return;
  const int e = idx >> 1, h = idx & 1;
  const int d = dst[e];
  const float m = ord2f(mx[(size_t)d * 2 + h]);
  const float ex = __expf(logit[idx] - m);
  exb[idx] = ex;
  atomicAdd(&den[(size_t)d * 2 + h], ex);
}

// Edge pass 3: agg[dst] += xl[src] * alpha.  One wave per edge.
__global__ void edge_scatter_kernel(const float* __restrict__ xl,
                                    const int* __restrict__ src,
                                    const int* __restrict__ dst,
                                    const float* __restrict__ exb,
                                    const float* __restrict__ den,
                                    float* __restrict__ agg, int E) {
  const int e = blockIdx.x * (blockDim.x >> 5) + (threadIdx.x >> 5);
  const int l = threadIdx.x & 31;
  if (e >= E) return;
  const int s = src[e], d = dst[e];
  const int h = l >> 4;
  const float alpha = exb[(size_t)e * 2 + h] / den[(size_t)d * 2 + h];
  const int cbase = l * 4;
#pragma unroll
  for (int j = 0; j < 4; ++j) {
    const int c = cbase + j;
    atomicAdd(&agg[(size_t)d * HD + c], xl[(size_t)s * HD + c] * alpha);
  }
}

// ---------------------------------------------------------------------------
// LayerNorm over 128 channels, one wave per row; optional residual/extra bias.
// ---------------------------------------------------------------------------
__global__ void ln_kernel(const float* __restrict__ in,
                          const float* __restrict__ res,
                          const float* __restrict__ addb,
                          const float* __restrict__ g,
                          const float* __restrict__ b,
                          float* __restrict__ out, int rows) {
  const int n = blockIdx.x * (blockDim.x >> 5) + (threadIdx.x >> 5);
  const int l = threadIdx.x & 31;
  if (n >= rows) return;
  float v[4];
  float s = 0.0f;
#pragma unroll
  for (int j = 0; j < 4; ++j) {
    const int c = l * 4 + j;
    float x = in[(size_t)n * HD + c];
    if (res)  x += res[(size_t)n * HD + c];
    if (addb) x += addb[c];
    v[j] = x;
    s += x;
  }
  for (int off = 1; off < 32; off <<= 1) s += __shfl_xor(s, off, 32);
  const float mean = s * (1.0f / HD);
  float vs = 0.0f;
#pragma unroll
  for (int j = 0; j < 4; ++j) { const float d0 = v[j] - mean; vs += d0 * d0; }
  for (int off = 1; off < 32; off <<= 1) vs += __shfl_xor(vs, off, 32);
  const float inv = rsqrtf(vs * (1.0f / HD) + 1e-5f);
#pragma unroll
  for (int j = 0; j < 4; ++j) {
    const int c = l * 4 + j;
    out[(size_t)n * HD + c] = (v[j] - mean) * inv * g[c] + b[c];
  }
}

// Per-graph mean/max/min pooling -> gfin[B, 3*HD]. One block per graph.
__global__ void pool_kernel(const float* __restrict__ h, float* __restrict__ gfin,
                            int per) {
  const int g = blockIdx.x, c = threadIdx.x;   // blockDim.x == HD
  float s = 0.0f, mx = -INFINITY, mn = INFINITY;
  const float* base = h + (size_t)g * per * HD + c;
  for (int i = 0; i < per; ++i) {
    const float v = base[(size_t)i * HD];
    s += v; mx = fmaxf(mx, v); mn = fminf(mn, v);
  }
  float* o = gfin + (size_t)g * (3 * HD);
  o[c] = s / (float)per;
  o[HD + c] = mx;
  o[2 * HD + c] = mn;
}

// logits[n] = dot(t[n,:], w[128]) + b0.  One wave per node.
__global__ void head_dot_kernel(const float* __restrict__ t,
                                const float* __restrict__ w,
                                const float* __restrict__ b0,
                                float* __restrict__ outv, int rows) {
  const int n = blockIdx.x * (blockDim.x >> 5) + (threadIdx.x >> 5);
  const int l = threadIdx.x & 31;
  if (n >= rows) return;
  float p = 0.0f;
#pragma unroll
  for (int j = 0; j < 4; ++j) {
    const int c = l * 4 + j;
    p += t[(size_t)n * HD + c] * w[c];
  }
  for (int off = 1; off < 32; off <<= 1) p += __shfl_xor(p, off, 32);
  if (l == 0) outv[n] = p + b0[0];
}

// Per-graph: argmax, log-softmax at argmax, entropy. One 256-thread block/graph.
__global__ void policy_kernel(const float* __restrict__ logits,
                              const unsigned char* __restrict__ mask,
                              float* __restrict__ out, int per, int B) {
  const int g = blockIdx.x, t = threadIdx.x;
  __shared__ float sv[256];
  __shared__ int   si[256];
  float best = -INFINITY; int bidx = 0;
  for (int i = t; i < per; i += 256) {
    float l = logits[(size_t)g * per + i];
    if (!mask[(size_t)g * per + i]) l = -INFINITY;
    if (l > best) { best = l; bidx = i; }
  }
  sv[t] = best; si[t] = bidx; __syncthreads();
  for (int s = 128; s > 0; s >>= 1) {
    if (t < s) {
      if (sv[t + s] > sv[t] || (sv[t + s] == sv[t] && si[t + s] < si[t])) {
        sv[t] = sv[t + s]; si[t] = si[t + s];
      }
    }
    __syncthreads();
  }
  const float mx = sv[0]; const int act = si[0]; __syncthreads();

  float s1 = 0.0f, s2 = 0.0f;
  for (int i = t; i < per; i += 256) {
    if (!mask[(size_t)g * per + i]) continue;
    const float d = logits[(size_t)g * per + i] - mx;
    const float e = __expf(d);
    s1 += e; s2 += d * e;
  }
  sv[t] = s1; __syncthreads();
  for (int s = 128; s > 0; s >>= 1) { if (t < s) sv[t] += sv[t + s]; __syncthreads(); }
  s1 = sv[0]; __syncthreads();
  sv[t] = s2; __syncthreads();
  for (int s = 128; s > 0; s >>= 1) { if (t < s) sv[t] += sv[t + s]; __syncthreads(); }
  s2 = sv[0];
  if (t == 0) {
    const float logZ = logf(s1);
    out[g]         = (float)act;       // actions
    out[B + g]     = -logZ;            // logprob of argmax (d_action == 0)
    out[2 * B + g] = logZ - s2 / s1;   // entropy
  }
}

// Critic: value[g] = relu(gf[g] @ cW1 + cb1) @ cW2 + cb2.  One block per graph.
__global__ void value_kernel(const float* __restrict__ gf,
                             const float* __restrict__ cW1,
                             const float* __restrict__ cb1,
                             const float* __restrict__ cW2,
                             const float* __restrict__ cb2,
                             float* __restrict__ out, int B) {
  const int g = blockIdx.x, j = threadIdx.x;   // blockDim.x == HD
  __shared__ float sv[HD];
  float acc = cb1[j];
  for (int c = 0; c < HD; ++c) acc += gf[(size_t)g * HD + c] * cW1[(size_t)c * HD + j];
  acc = fmaxf(acc, 0.0f);
  sv[j] = acc * cW2[j]; __syncthreads();
  for (int s = HD / 2; s > 0; s >>= 1) { if (j < s) sv[j] += sv[j + s]; __syncthreads(); }
  if (j == 0) out[3 * B + g] = sv[0] + cb2[0];
}

// ---------------------------------------------------------------------------
extern "C" void kernel_launch(void* const* d_in, const int* in_sizes, int n_in,
                              void* d_out, int out_size, void* d_ws, size_t ws_size,
                              hipStream_t stream) {
  (void)in_sizes; (void)n_in; (void)out_size; (void)ws_size;
  const int N = NNODES, E = NEDGES, B = NGRAPH, per = N / B;

  const float* x     = (const float*)d_in[0];
  const float* eattr = (const float*)d_in[1];
  const int*   eidx  = (const int*)d_in[2];
  const unsigned char* mask = (const unsigned char*)d_in[3];
  const float* Wn  = (const float*)d_in[4];   const float* bn  = (const float*)d_in[5];
  const float* Wep = (const float*)d_in[6];   const float* bep = (const float*)d_in[7];
  const float* gWl = (const float*)d_in[8];   const float* gbl = (const float*)d_in[9];
  const float* gWr = (const float*)d_in[10];  const float* gbr = (const float*)d_in[11];
  const float* gWe = (const float*)d_in[12];  const float* gatt = (const float*)d_in[13];
  const float* gbias = (const float*)d_in[14];
  const float* ln_g = (const float*)d_in[15]; const float* ln_b = (const float*)d_in[16];
  const float* gnW = (const float*)d_in[17];  const float* gnb = (const float*)d_in[18];
  const float* gn_g = (const float*)d_in[19]; const float* gn_b = (const float*)d_in[20];
  // d_in[21..24] = edge_net params: dead code in reference (never returned)
  const float* aW1 = (const float*)d_in[25];  const float* ab1 = (const float*)d_in[26];
  const float* aW2 = (const float*)d_in[27];  const float* ab2 = (const float*)d_in[28];
  const float* cW1 = (const float*)d_in[29];  const float* cb1 = (const float*)d_in[30];
  const float* cW2 = (const float*)d_in[31];  const float* cb2 = (const float*)d_in[32];

  const int* srcI = eidx;
  const int* dstI = eidx + E;

  // ---- workspace carving (256B aligned chunks) ----
  char* base = (char*)d_ws;
  size_t off = 0;
  auto alloc = [&](size_t bytes) {
    off = (off + 255) & ~(size_t)255;
    void* p = base + off;
    off += bytes;
    return p;
  };
  float* h    = (float*)alloc((size_t)N * HD * 4);
  float* ebuf = (float*)alloc((size_t)E * HD * 4);
  float* xl   = (float*)alloc((size_t)N * HD * 4);
  float* xr   = (float*)alloc((size_t)N * HD * 4);
  float* xe   = (float*)alloc((size_t)E * HD * 4);
  float* lg   = (float*)alloc((size_t)E * 2 * 4);
  float* exb  = (float*)alloc((size_t)E * 2 * 4);
  unsigned* mxb = (unsigned*)alloc((size_t)N * 2 * 4);
  float* den  = (float*)alloc((size_t)N * 2 * 4);
  float* agg  = (float*)alloc((size_t)N * HD * 4);
  float* gfin = (float*)alloc((size_t)B * 3 * HD * 4);
  float* gfb  = (float*)alloc((size_t)B * HD * 4);
  float* lgts = (float*)alloc((size_t)N * 4);
  // bf16 operand copies
  __bf16* xbf   = (__bf16*)alloc((size_t)N * NF * 2);
  __bf16* eabf  = (__bf16*)alloc((size_t)E * EF * 2);
  __bf16* hbf   = (__bf16*)alloc((size_t)N * HD * 2);
  __bf16* ebbf  = (__bf16*)alloc((size_t)E * HD * 2);
  __bf16* gfinbf= (__bf16*)alloc((size_t)B * 3 * HD * 2);
  // transposed bf16 weights
  __bf16* WnT  = (__bf16*)alloc((size_t)HD * NF * 2);
  __bf16* WepT = (__bf16*)alloc((size_t)HD * EF * 2);
  __bf16* WlT  = (__bf16*)alloc((size_t)NLAYERS * HD * HD * 2);
  __bf16* WrT  = (__bf16*)alloc((size_t)NLAYERS * HD * HD * 2);
  __bf16* WeT  = (__bf16*)alloc((size_t)NLAYERS * HD * HD * 2);
  __bf16* gnWT = (__bf16*)alloc((size_t)HD * 3 * HD * 2);
  __bf16* aW1T = (__bf16*)alloc((size_t)HD * HD * 2);

  auto cvt = [&](const float* in, __bf16* out, long long n) {
    hipLaunchKernelGGL(f2bf_kernel, dim3(2048), dim3(256), 0, stream, in, out, n);
  };
  auto wpack = [&](const float* W, __bf16* Wt, int K) {
    hipLaunchKernelGGL(wtrans_kernel, dim3((K * HD + 255) / 256), dim3(256), 0,
                       stream, W, Wt, K);
  };

  // ---- pack all weights, convert inputs ----
  wpack(Wn, WnT, NF);
  wpack(Wep, WepT, EF);
  for (int i = 0; i < NLAYERS; ++i) {
    wpack(gWl + (size_t)i * HD * HD, WlT + (size_t)i * HD * HD, HD);
    wpack(gWr + (size_t)i * HD * HD, WrT + (size_t)i * HD * HD, HD);
    wpack(gWe + (size_t)i * HD * HD, WeT + (size_t)i * HD * HD, HD);
  }
  wpack(gnW, gnWT, 3 * HD);
  wpack(aW1, aW1T, HD);
  cvt(x, xbf, (long long)N * NF);
  cvt(eattr, eabf, (long long)E * EF);

  // ---- preprocessing MLPs (relu) ----
  hipLaunchKernelGGL(wmma_gemm_bf16<NF>, dim3(N / 128), dim3(256), 0, stream,
                     xbf, WnT, bn, h, 1);
  hipLaunchKernelGGL(wmma_gemm_bf16<EF>, dim3(E / 128), dim3(256), 0, stream,
                     eabf, WepT, bep, ebuf, 1);
  cvt(ebuf, ebbf, (long long)E * HD);   // e is constant across layers
  cvt(h, hbf, (long long)N * HD);

  const int edgeWaveBlocks = E / 8;     // 8 waves of 32 per 256-thread block
  for (int i = 0; i < NLAYERS; ++i) {
    hipLaunchKernelGGL(wmma_gemm_bf16<HD>, dim3(N / 128), dim3(256), 0, stream,
                       hbf, WlT + (size_t)i * HD * HD, gbl + i * HD, xl, 0);
    hipLaunchKernelGGL(wmma_gemm_bf16<HD>, dim3(N / 128), dim3(256), 0, stream,
                       hbf, WrT + (size_t)i * HD * HD, gbr + i * HD, xr, 0);
    hipLaunchKernelGGL(wmma_gemm_bf16<HD>, dim3(E / 128), dim3(256), 0, stream,
                       ebbf, WeT + (size_t)i * HD * HD, nullptr, xe, 0);

    hipLaunchKernelGGL(fill_kernel, dim3(512), dim3(256), 0, stream,
                       (float*)mxb, 0.0f, (long long)N * 2);  // ord() >= 0 always
    hipLaunchKernelGGL(fill_kernel, dim3(512), dim3(256), 0, stream,
                       den, 0.0f, (long long)N * 2);
    hipLaunchKernelGGL(fill_kernel, dim3(2048), dim3(256), 0, stream,
                       agg, 0.0f, (long long)N * HD);

    hipLaunchKernelGGL(edge_logit_kernel, dim3(edgeWaveBlocks), dim3(256), 0, stream,
                       xl, xr, xe, srcI, dstI, gatt + (size_t)i * HD, lg, mxb, E);
    hipLaunchKernelGGL(edge_exp_kernel, dim3((2 * E) / 256), dim3(256), 0, stream,
                       lg, dstI, mxb, exb, den, 2 * E);
    hipLaunchKernelGGL(edge_scatter_kernel, dim3(edgeWaveBlocks), dim3(256), 0, stream,
                       xl, srcI, dstI, exb, den, agg, E);

    // h = LN(h + agg + gbias); refresh bf16 copy of h
    hipLaunchKernelGGL(ln_kernel, dim3(N / 8), dim3(256), 0, stream,
                       h, agg, gbias + i * HD, ln_g + i * HD, ln_b + i * HD, h, N);
    cvt(h, hbf, (long long)N * HD);
  }

  // ---- global features: pooling -> relu GEMM -> LN ----
  hipLaunchKernelGGL(pool_kernel, dim3(B), dim3(HD), 0, stream, h, gfin, per);
  cvt(gfin, gfinbf, (long long)B * 3 * HD);
  hipLaunchKernelGGL(wmma_gemm_bf16<3 * HD>, dim3(1), dim3(32), 0, stream,
                     gfinbf, gnWT, gnb, gfb, 1);
  hipLaunchKernelGGL(ln_kernel, dim3((B + 7) / 8), dim3(256), 0, stream,
                     gfb, nullptr, nullptr, gn_g, gn_b, gfb, B);

  // ---- action head: t = relu(h@aW1+ab1) (reuse xl); logits = t@aW2+ab2 ----
  hipLaunchKernelGGL(wmma_gemm_bf16<HD>, dim3(N / 128), dim3(256), 0, stream,
                     hbf, aW1T, ab1, xl, 1);
  hipLaunchKernelGGL(head_dot_kernel, dim3(N / 8), dim3(256), 0, stream,
                     xl, aW2, ab2, lgts, N);

  float* out = (float*)d_out;   // [actions | logprobs | entropy | value], B each
  hipLaunchKernelGGL(policy_kernel, dim3(B), dim3(256), 0, stream,
                     lgts, mask, out, per, B);
  hipLaunchKernelGGL(value_kernel, dim3(B), dim3(HD), 0, stream,
                     gfb, cW1, cb1, cW2, cb2, out, B);
}
